// embed_39376260170350
// MI455X (gfx1250) — matrix operand order repaired and checked by reading, not compiled, not run on hardware
//
#include <hip/hip_runtime.h>

// ---------------------------------------------------------------------------
// out = inputs @ W + b   (M=2048, K=50257, N=128, fp32)
// bf16x2-split GEMM on v_wmma_f32_16x16x32_bf16 (3 WMMAs per tile ~ fp32
// accuracy). Split-K=16 -> 256 WGs x 8 wave32. Software-pipelined: register
// prefetch of step s+2 overlaps the WMMA phase of step s; double-buffered LDS
// (80 KB of the 320 KB WGP pool) gives one barrier per K-step.
// Roofline: 439 MB @ 23.3 TB/s ~= 19 us floor; 3x26.4=79 GFLOP bf16 ~= 16 us.
// ---------------------------------------------------------------------------

typedef __attribute__((ext_vector_type(16))) __bf16 v16bf;
typedef __attribute__((ext_vector_type(8)))  float  v8f;

#define VOCAB   50257
#define NEMB    128
#define MROWS   2048          // 16 * 128
#define MT      16            // number of 128-row M tiles
#define SPLITK  16
#define KCHUNK  3142          // ceil(50257 / 16)
#define KT      32            // K per WMMA step
#define ROWU32  20            // LDS row stride in u32 (40 bf16 = 80 B)

static __device__ __forceinline__ void split_pack(float x0, float x1,
                                                  unsigned& hi, unsigned& lo) {
  __bf16 h0 = (__bf16)x0, h1 = (__bf16)x1;
  __bf16 l0 = (__bf16)(x0 - (float)h0);
  __bf16 l1 = (__bf16)(x1 - (float)h1);
  union { __bf16 b[2]; unsigned u; } ph, pl;
  ph.b[0] = h0; ph.b[1] = h1;
  pl.b[0] = l0; pl.b[1] = l1;
  hi = ph.u; lo = pl.u;
}

// MODE 0: store partial tile to ws[ks][row][col]; MODE 1: atomicAdd into out.
template <int MODE>
__global__ __launch_bounds__(256) void gemm_bf16x3(const float* __restrict__ A,
                                                   const float* __restrict__ W,
                                                   float* __restrict__ outp) {
  __shared__ __align__(16) unsigned sAh[2][128 * ROWU32];
  __shared__ __align__(16) unsigned sAl[2][128 * ROWU32];
  __shared__ __align__(16) unsigned sWh[2][128 * ROWU32];
  __shared__ __align__(16) unsigned sWl[2][128 * ROWU32];

  const int tid  = threadIdx.x;
  const int mt   = blockIdx.x & (MT - 1);
  const int ks   = blockIdx.x >> 4;
  const int k0   = ks * KCHUNK;
  const int k1   = (k0 + KCHUNK < VOCAB) ? (k0 + KCHUNK) : VOCAB;
  const int mtrow = mt * 128;

  const int lane = tid & 31;
  const int wv   = tid >> 5;     // wave 0..7 -> M strip wv*16
  const int half = lane >> 4;    // fragment half (ISA 7.12.2 layouts)
  const int l16  = lane & 15;

  // Per-thread staging coordinates (fixed across steps).
  const int ar[8]  = { (tid + 0*256) >> 4, (tid + 1*256) >> 4, (tid + 2*256) >> 4,
                       (tid + 3*256) >> 4, (tid + 4*256) >> 4, (tid + 5*256) >> 4,
                       (tid + 6*256) >> 4, (tid + 7*256) >> 4 };
  const int ac2    = tid & 15;            // same for every it (idx & 15)
  const int wn     = tid & 127;           // W: n column (idx & 127)
  const int wc2[8] = { (tid + 0*256) >> 7, (tid + 1*256) >> 7, (tid + 2*256) >> 7,
                       (tid + 3*256) >> 7, (tid + 4*256) >> 7, (tid + 5*256) >> 7,
                       (tid + 6*256) >> 7, (tid + 7*256) >> 7 };

  v8f acc[8];
#pragma unroll
  for (int i = 0; i < 8; ++i) acc[i] = 0.0f;

  const int steps = (k1 - k0 + KT - 1) / KT;

  float ra[16];   // prefetched A f32 (8 pairs)
  float rw[16];   // prefetched W f32 (8 pairs)

  // ---- global load of one KT tile into registers (fast path / padded tail) --
  auto gload = [&](int s) {
    const int kb = k0 + s * KT;
    if (kb + KT <= k1) {
#pragma unroll
      for (int it = 0; it < 8; ++it) {
        size_t ba = (size_t)(mtrow + ar[it]) * VOCAB + (kb + 2 * ac2);
        ra[2 * it]     = A[ba];
        ra[2 * it + 1] = A[ba + 1];
        size_t bw = (size_t)(kb + 2 * wc2[it]) * NEMB + wn;
        rw[2 * it]     = W[bw];
        rw[2 * it + 1] = W[bw + NEMB];
      }
    } else {
#pragma unroll
      for (int it = 0; it < 8; ++it) {
        int kA = kb + 2 * ac2;
        size_t ba = (size_t)(mtrow + ar[it]) * VOCAB + kA;
        ra[2 * it]     = (kA     < k1) ? A[ba]     : 0.0f;
        ra[2 * it + 1] = (kA + 1 < k1) ? A[ba + 1] : 0.0f;
        int kW = kb + 2 * wc2[it];
        size_t bw = (size_t)kW * NEMB + wn;
        rw[2 * it]     = (kW     < k1) ? W[bw]        : 0.0f;
        rw[2 * it + 1] = (kW + 1 < k1) ? W[bw + NEMB] : 0.0f;
      }
    }
  };

  // ---- convert prefetched registers to bf16 hi/lo and store into buffer p --
  auto cstore = [&](int p) {
#pragma unroll
    for (int it = 0; it < 8; ++it) {
      unsigned hi, lo;
      split_pack(ra[2 * it], ra[2 * it + 1], hi, lo);
      sAh[p][ar[it] * ROWU32 + ac2] = hi;
      sAl[p][ar[it] * ROWU32 + ac2] = lo;
      split_pack(rw[2 * it], rw[2 * it + 1], hi, lo);
      sWh[p][wn * ROWU32 + wc2[it]] = hi;
      sWl[p][wn * ROWU32 + wc2[it]] = lo;
    }
  };

  // ---- WMMA phase from buffer p ----
  const int arow  = wv * 16 + l16;
  const int aoff0 = arow * ROWU32 + half * 4;   // A: K 0..7 / 8..15
  const int aoff1 = aoff0 + 8;                  //    K 16..23 / 24..31
  auto compute = [&](int p) {
    v16bf ah, al;
    ((uint4*)&ah)[0] = *(const uint4*)&sAh[p][aoff0];
    ((uint4*)&ah)[1] = *(const uint4*)&sAh[p][aoff1];
    ((uint4*)&al)[0] = *(const uint4*)&sAl[p][aoff0];
    ((uint4*)&al)[1] = *(const uint4*)&sAl[p][aoff1];
#pragma unroll
    for (int nt = 0; nt < 8; ++nt) {
      const int ncol = nt * 16 + l16;           // B: lane holds column n
      const int woff = ncol * ROWU32 + half * 4;
      v16bf wh, wl;
      ((uint4*)&wh)[0] = *(const uint4*)&sWh[p][woff];
      ((uint4*)&wh)[1] = *(const uint4*)&sWh[p][woff + 8];
      ((uint4*)&wl)[0] = *(const uint4*)&sWl[p][woff];
      ((uint4*)&wl)[1] = *(const uint4*)&sWl[p][woff + 8];
      // (Ah+Al)(Wh+Wl) ~= Ah*Wh + Ah*Wl + Al*Wh
      acc[nt] = __builtin_amdgcn_wmma_f32_16x16x32_bf16(
          false, ah, false, wh, (short)0, acc[nt], false, false);
      acc[nt] = __builtin_amdgcn_wmma_f32_16x16x32_bf16(
          false, ah, false, wl, (short)0, acc[nt], false, false);
      acc[nt] = __builtin_amdgcn_wmma_f32_16x16x32_bf16(
          false, al, false, wh, (short)0, acc[nt], false, false);
    }
  };

  // ---- software pipeline: prefetch s+2 overlaps WMMA of s ----
  gload(0);
  cstore(0);
  if (steps > 1) gload(1);
  __syncthreads();

  for (int s = 0; s < steps; ++s) {
    const int p = s & 1;
    if (s + 1 < steps) cstore(1 - p);   // regs from iteration s-1 (landed)
    if (s + 2 < steps) gload(s + 2);    // in flight during this WMMA phase
    compute(p);
    __syncthreads();
  }

  // ---- writeback: C/D layout -> row = half*8 + j, col = lane%16 ----
  const int rbase = mtrow + wv * 16 + half * 8;
#pragma unroll
  for (int nt = 0; nt < 8; ++nt) {
    const int col = nt * 16 + l16;
#pragma unroll
    for (int j = 0; j < 8; ++j) {
      const int row = rbase + j;
      if (MODE == 1) {
        atomicAdd(&outp[(size_t)row * NEMB + col], acc[nt][j]);
      } else {
        outp[((size_t)ks * MROWS + row) * NEMB + col] = acc[nt][j];
      }
    }
  }
}

__global__ __launch_bounds__(256) void reduce_bias(const float* __restrict__ ws,
                                                   const float* __restrict__ b,
                                                   float* __restrict__ out) {
  const int i = blockIdx.x * 256 + threadIdx.x;   // 0 .. 2048*128-1
  float s = b[i & (NEMB - 1)];
#pragma unroll
  for (int k = 0; k < SPLITK; ++k)
    s += ws[(size_t)k * MROWS * NEMB + i];
  out[i] = s;
}

__global__ __launch_bounds__(256) void bias_init(const float* __restrict__ b,
                                                 float* __restrict__ out) {
  const int i = blockIdx.x * 256 + threadIdx.x;
  out[i] = b[i & (NEMB - 1)];
}

extern "C" void kernel_launch(void* const* d_in, const int* in_sizes, int n_in,
                              void* d_out, int out_size, void* d_ws, size_t ws_size,
                              hipStream_t stream) {
  const float* A = (const float*)d_in[0];   // (16,128,50257) fp32
  const float* W = (const float*)d_in[1];   // (50257,128) fp32
  const float* b = (const float*)d_in[2];   // (128,) fp32
  float* out = (float*)d_out;               // (16,128,128) fp32

  const int nOut = MROWS * NEMB;            // 262144
  const size_t need = (size_t)SPLITK * MROWS * NEMB * sizeof(float); // 16 MB

  if (d_ws != nullptr && ws_size >= need) {
    // Deterministic path: split-K partials in workspace, then reduce + bias.
    gemm_bf16x3<0><<<dim3(MT * SPLITK), dim3(256), 0, stream>>>(A, W, (float*)d_ws);
    reduce_bias<<<dim3(nOut / 256), dim3(256), 0, stream>>>((const float*)d_ws, b, out);
  } else {
    // Fallback: bias-init output, accumulate splits with global f32 atomics.
    bias_init<<<dim3(nOut / 256), dim3(256), 0, stream>>>(b, out);
    gemm_bf16x3<1><<<dim3(MT * SPLITK), dim3(256), 0, stream>>>(A, W, out);
  }
}
